// HorNet_74586402062553
// MI455X (gfx1250) — matrix operand order repaired and checked by reading, not compiled, not run on hardware
//
#include <hip/hip_runtime.h>

// MI455X / gfx1250 fused HorNet kernel.
// Per block (256 threads = 8 wave32): 32 rows.
//  P0: pack x rows (binary) into 32-bit masks
//  P1: scores via bit-test selects, pow9, clip -> LDS
//  P2: softmax over R=256 (8 lanes/row, shfl reductions), attn -> f16 LDS (WMMA-A layout friendly)
//  P3: v_wmma_f32_16x16x32_f16: attn(32x256) @ w_outT(256x128pad), B preloaded per wave
//  P4: log_softmax over OUT=100, store

typedef _Float16 v16h  __attribute__((ext_vector_type(16)));
typedef _Float16 f16x8 __attribute__((ext_vector_type(8)));
typedef float    v8f   __attribute__((ext_vector_type(8)));

#define N_F     32
#define N_R     256
#define N_O     5
#define N_OUT   100
#define ROWS    32
#define SC_STR  264   // f32 score stride (264%64==8 -> conflict-free softmax passes)
#define AT_STR  264   // f16 attn stride  (528B row -> 4-bank rotation, conflict-free b128 A loads)
#define OB_STR  132   // f32 out stride   (132%64==4 -> lane groups hit disjoint banks)

__global__ __launch_bounds__(256)
void hornet_fused(const float* __restrict__ x,
                  const float* __restrict__ comb_space,
                  const float* __restrict__ w_out,
                  const float* __restrict__ b_out,
                  const int*   __restrict__ comb_idx,
                  float*       __restrict__ out)
{
    __shared__ __align__(16) float    s_scores[ROWS * SC_STR];   // 33792 B (aliased as out_buf in P3/P4)
    __shared__ __align__(16) _Float16 s_attn  [ROWS * AT_STR];   // 16896 B
    __shared__ unsigned int           s_mask  [ROWS];

    const int t  = threadIdx.x;
    const int n0 = blockIdx.x * ROWS;

    // ---- P0: pack binary feature rows into bitmasks (F == 32) ----
    if (t < ROWS) {
        const float4* xp = (const float4*)(x + (size_t)(n0 + t) * N_F);
        unsigned int m = 0u;
        #pragma unroll
        for (int i = 0; i < 8; ++i) {
            float4 v = xp[i];
            m |= (v.x > 0.5f ? 1u : 0u) << (4 * i + 0);
            m |= (v.y > 0.5f ? 1u : 0u) << (4 * i + 1);
            m |= (v.z > 0.5f ? 1u : 0u) << (4 * i + 2);
            m |= (v.w > 0.5f ? 1u : 0u) << (4 * i + 3);
        }
        s_mask[t] = m;
    }

    // per-thread combination data (thread == r)
    int   ci[N_O];
    float cw[N_O];
    #pragma unroll
    for (int o = 0; o < N_O; ++o) {
        ci[o] = comb_idx[t * N_O + o];
        cw[o] = comb_space[o * N_R + t];
    }
    __syncthreads();

    // ---- P1: scores -> pow9 -> clip ----
    #pragma unroll 4
    for (int row = 0; row < ROWS; ++row) {
        const unsigned int m = s_mask[row];
        float sc = 0.f;
        #pragma unroll
        for (int o = 0; o < N_O; ++o)
            sc += ((m >> ci[o]) & 1u) ? cw[o] : 0.f;
        const float s2 = sc * sc, s4 = s2 * s2, s8 = s4 * s4;
        float p = s8 * sc;                       // sc^9 (odd power, sign kept)
        p = fminf(fmaxf(p, -1.f), 1.f);
        s_scores[row * SC_STR + t] = p;
    }
    __syncthreads();

    // ---- P2: softmax over R per row; attn -> f16 LDS ----
    {
        const int row = t >> 3;
        const int l   = t & 7;
        float mx = -__builtin_inff();
        #pragma unroll 8
        for (int k = 0; k < 32; ++k)
            mx = fmaxf(mx, s_scores[row * SC_STR + l + 8 * k]);
        mx = fmaxf(mx, __shfl_xor(mx, 1, 8));
        mx = fmaxf(mx, __shfl_xor(mx, 2, 8));
        mx = fmaxf(mx, __shfl_xor(mx, 4, 8));
        float sum = 0.f;
        #pragma unroll 8
        for (int k = 0; k < 32; ++k) {
            const int o = row * SC_STR + l + 8 * k;
            const float e = __expf(s_scores[o] - mx);
            s_scores[o] = e;
            sum += e;
        }
        sum += __shfl_xor(sum, 1, 8);
        sum += __shfl_xor(sum, 2, 8);
        sum += __shfl_xor(sum, 4, 8);
        const float inv = 1.0f / sum;
        #pragma unroll 8
        for (int k = 0; k < 32; ++k) {
            const int c = l + 8 * k;
            s_attn[row * AT_STR + c] =
                (_Float16)(s_scores[row * SC_STR + c] * inv);
        }
    }
    __syncthreads();

    // ---- P3: WMMA GEMM  D(32x128) = attn(32x256) x w_outT(256x128pad) + b_out ----
    const int wv     = t >> 5;
    const int lane   = t & 31;
    const int lane16 = lane & 15;
    const int hig    = lane >> 4;            // lane group 0/1
    const int c      = wv * 16 + lane16;     // output column this lane owns (B/D: N = lane%16)
    const int c_safe = (c < N_OUT) ? c : 0;  // padded cols: garbage B only pollutes unsaved lanes

    // Preload B panel (32x16 f16), contiguous k order: V0..7, lanes0-15 hold K=0..15, lanes16-31 K=16..31.
    // -> lane needs 16 contiguous K values of column c: w_out[c*256 + kb + 16*hig .. +15]
    v16h B[8];
    #pragma unroll
    for (int s = 0; s < 8; ++s) {
        const float4* bp = (const float4*)(w_out + (size_t)c_safe * N_R + 32 * s + 16 * hig);
        #pragma unroll
        for (int q = 0; q < 4; ++q) {
            const float4 v = bp[q];
            B[s][4 * q + 0] = (_Float16)v.x;
            B[s][4 * q + 1] = (_Float16)v.y;
            B[s][4 * q + 2] = (_Float16)v.z;
            B[s][4 * q + 3] = (_Float16)v.w;
        }
    }

    v8f acc0, acc1;
    const float bv = b_out[c_safe];
    #pragma unroll
    for (int i = 0; i < 8; ++i) { acc0[i] = bv; acc1[i] = bv; }

    // A layout (16x32 f16), matching B: lanes0-15 hold K=0..15, lanes16-31 K=16..31.
    const int ka = 16 * hig;
    #pragma unroll
    for (int s = 0; s < 8; ++s) {
        const int kb = 32 * s;
        {   // row-tile 0: rows 0..15
            const _Float16* ap = &s_attn[lane16 * AT_STR + kb + ka];
            const f16x8 lo = *(const f16x8*)ap;
            const f16x8 hi = *(const f16x8*)(ap + 8);
            v16h a;
            #pragma unroll
            for (int e = 0; e < 8; ++e) { a[e] = lo[e]; a[8 + e] = hi[e]; }
            acc0 = __builtin_amdgcn_wmma_f32_16x16x32_f16(
                       false, a, false, B[s], (short)0, acc0, false, false);
        }
        {   // row-tile 1: rows 16..31
            const _Float16* ap = &s_attn[(16 + lane16) * AT_STR + kb + ka];
            const f16x8 lo = *(const f16x8*)ap;
            const f16x8 hi = *(const f16x8*)(ap + 8);
            v16h a;
            #pragma unroll
            for (int e = 0; e < 8; ++e) { a[e] = lo[e]; a[8 + e] = hi[e]; }
            acc1 = __builtin_amdgcn_wmma_f32_16x16x32_f16(
                       false, a, false, B[s], (short)0, acc1, false, false);
        }
    }

    // D layout: VGPR i: lanes0-15 -> M=i, lanes16-31 -> M=i+8, N=lane%16.
    float* ob = s_scores;  // scores buffer is dead -> reuse as out_buf[32][OB_STR]
    if (c < N_OUT) {
        #pragma unroll
        for (int i = 0; i < 8; ++i) {
            const int rd = 8 * hig + i;
            ob[rd        * OB_STR + c] = acc0[i];
            ob[(16 + rd) * OB_STR + c] = acc1[i];
        }
    }
    __syncthreads();

    // ---- P4: log_softmax over OUT=100 and store ----
    {
        const int row = t >> 3;
        const int l   = t & 7;
        float mx = -__builtin_inff();
        #pragma unroll
        for (int k = 0; k < 13; ++k) {
            const int cc = l + 8 * k;
            if (cc < N_OUT) mx = fmaxf(mx, ob[row * OB_STR + cc]);
        }
        mx = fmaxf(mx, __shfl_xor(mx, 1, 8));
        mx = fmaxf(mx, __shfl_xor(mx, 2, 8));
        mx = fmaxf(mx, __shfl_xor(mx, 4, 8));
        float sum = 0.f;
        #pragma unroll
        for (int k = 0; k < 13; ++k) {
            const int cc = l + 8 * k;
            if (cc < N_OUT) sum += __expf(ob[row * OB_STR + cc] - mx);
        }
        sum += __shfl_xor(sum, 1, 8);
        sum += __shfl_xor(sum, 2, 8);
        sum += __shfl_xor(sum, 4, 8);
        const float lse = __logf(sum) + mx;
        float* op = out + (size_t)(n0 + row) * N_OUT;
        #pragma unroll
        for (int k = 0; k < 13; ++k) {
            const int cc = l + 8 * k;
            if (cc < N_OUT) op[cc] = ob[row * OB_STR + cc] - lse;
        }
    }
}

extern "C" void kernel_launch(void* const* d_in, const int* in_sizes, int n_in,
                              void* d_out, int out_size, void* d_ws, size_t ws_size,
                              hipStream_t stream) {
    const float* x          = (const float*)d_in[0];
    const float* comb_space = (const float*)d_in[1];
    const float* w_out      = (const float*)d_in[2];
    const float* b_out      = (const float*)d_in[3];
    const int*   comb_idx   = (const int*)d_in[4];
    float*       out        = (float*)d_out;

    const int n = in_sizes[0] / N_F;          // N rows (131072)
    const int grid = n / ROWS;                // 4096 blocks
    hornet_fused<<<grid, 256, 0, stream>>>(x, comb_space, w_out, b_out, comb_idx, out);
    (void)d_ws; (void)ws_size; (void)n_in; (void)out_size;
}
